// MultiHeadSelfAttn_64982855188496
// MI455X (gfx1250) — matrix operand order repaired and checked by reading, not compiled, not run on hardware
//
#include <hip/hip_runtime.h>
#include <hip/hip_bf16.h>
#include <math.h>

typedef __attribute__((ext_vector_type(16))) __bf16 v16bf;
typedef __attribute__((ext_vector_type(8)))  __bf16 v8bf;
typedef __attribute__((ext_vector_type(8)))  float  v8f;

#define D_MODEL 1024
#define N_HEADS 16
#define DK      64
#define SEQ     2048
#define BATCH   2
#define NTOK    (BATCH * SEQ)      // 4096
#define QKV_N   (3 * D_MODEL)      // 3072

static __device__ inline v8f wmma_bf16(v16bf a, v16bf b, v8f c) {
  return __builtin_amdgcn_wmma_f32_16x16x32_bf16(
      /*neg_a=*/false, a, /*neg_b=*/false, b,
      /*c_mod=*/(short)0, c, /*reuse_a=*/false, /*reuse_b=*/false);
}

// ---------------------------------------------------------------------------
// xor-style reductions within each 16-lane half via v_permlane16_b32 (VALU,
// no LDS pipe). Selector nibbles: dest lane i reads source lane sel[i].
// ---------------------------------------------------------------------------
#if __has_builtin(__builtin_amdgcn_permlane16)
static __device__ inline float plane16(float v, unsigned lo, unsigned hi) {
  int s = __float_as_int(v);
  int r = __builtin_amdgcn_permlane16(s, s, (int)lo, (int)hi, false, false);
  return __int_as_float(r);
}
static __device__ inline float redmax16(float v) {
  v = fmaxf(v, plane16(v, 0x67452301u, 0xEFCDAB89u));  // xor 1
  v = fmaxf(v, plane16(v, 0x54761032u, 0xDCFE98BAu));  // xor 2
  v = fmaxf(v, plane16(v, 0x32107654u, 0xBA98FEDCu));  // xor 4
  v = fmaxf(v, plane16(v, 0xFEDCBA98u, 0x76543210u));  // xor 8
  return v;
}
static __device__ inline float redsum16(float v) {
  v += plane16(v, 0x67452301u, 0xEFCDAB89u);
  v += plane16(v, 0x54761032u, 0xDCFE98BAu);
  v += plane16(v, 0x32107654u, 0xBA98FEDCu);
  v += plane16(v, 0xFEDCBA98u, 0x76543210u);
  return v;
}
#else
static __device__ inline float redmax16(float v) {
  for (int m = 1; m < 16; m <<= 1) v = fmaxf(v, __shfl_xor(v, m, 32));
  return v;
}
static __device__ inline float redsum16(float v) {
  for (int m = 1; m < 16; m <<= 1) v += __shfl_xor(v, m, 32);
  return v;
}
#endif

// ---------------------------------------------------------------------------
// fp32 -> bf16 conversion
// ---------------------------------------------------------------------------
__global__ void cvt_f32_bf16_kernel(const float* __restrict__ in,
                                    __bf16* __restrict__ out, int n) {
  int i = blockIdx.x * blockDim.x + threadIdx.x;
  if (i < n) out[i] = (__bf16)in[i];
}

// ---------------------------------------------------------------------------
// C[M,N] = A[M,K] @ B[N,K]^T   (torch Linear convention; A,B bf16 row-major)
// Block: 256 threads (8 waves), 128x128 tile; each wave: 32(M) x 64(N).
// ---------------------------------------------------------------------------
template <bool OUT_BF16>
__global__ __launch_bounds__(256)
void wmma_gemm_kernel(const __bf16* __restrict__ A, const __bf16* __restrict__ B,
                      void* __restrict__ Cout, const float* __restrict__ bias,
                      int M, int N, int K) {
  const int lane = threadIdx.x & 31;
  const int wave = threadIdx.x >> 5;
  const int wm = wave & 3;           // 0..3 -> M sub-tile
  const int wn = wave >> 2;          // 0..1 -> N sub-tile
  const int m0 = blockIdx.y * 128 + wm * 32;
  const int n0 = blockIdx.x * 128 + wn * 64;
  const int l15 = lane & 15;
  const int hiHalf = lane >> 4;
  const int koA = hiHalf ? 16 : 0;   // A-frag K offset per lane half
  const int koB = hiHalf ? 16 : 0;   // B-frag K offset per lane half

  v8f acc[2][4] = {};

  const __bf16* Arow[2];
  Arow[0] = A + (size_t)(m0 + l15) * K;
  Arow[1] = Arow[0] + (size_t)16 * K;

#pragma unroll 2
  for (int k0 = 0; k0 < K; k0 += 32) {
    v16bf a[2], b[4];
#pragma unroll
    for (int i = 0; i < 2; ++i) {
      const __bf16* p = Arow[i] + k0 + koA;
      ((v8bf*)&a[i])[0] = *(const v8bf*)(p);
      ((v8bf*)&a[i])[1] = *(const v8bf*)(p + 8);
    }
#pragma unroll
    for (int j = 0; j < 4; ++j) {
      const __bf16* p = B + (size_t)(n0 + j * 16 + l15) * K + k0 + koB;
      ((v8bf*)&b[j])[0] = *(const v8bf*)(p);
      ((v8bf*)&b[j])[1] = *(const v8bf*)(p + 8);
    }
#pragma unroll
    for (int i = 0; i < 2; ++i)
#pragma unroll
      for (int j = 0; j < 4; ++j)
        acc[i][j] = wmma_bf16(a[i], b[j], acc[i][j]);
  }

  // Epilogue: C/D layout -> VGPR r = row r (+8 upper lanes), lane = col.
#pragma unroll
  for (int i = 0; i < 2; ++i) {
#pragma unroll
    for (int r = 0; r < 8; ++r) {
      const int row = m0 + i * 16 + r + (hiHalf ? 8 : 0);
#pragma unroll
      for (int j = 0; j < 4; ++j) {
        const int col = n0 + j * 16 + l15;
        const size_t idx = (size_t)row * N + col;
        if (OUT_BF16) {
          ((__bf16*)Cout)[idx] = (__bf16)acc[i][j][r];
        } else {
          float v = acc[i][j][r];
          if (bias) v += bias[col];
          ((float*)Cout)[idx] = v;
        }
      }
    }
  }
}

// ---------------------------------------------------------------------------
// Flash attention over one (batch, head): Q,K,V are 64-wide slices of the
// bf16 qkv buffer [NTOK, 3*D_MODEL]. Block = 128 threads (4 waves), 128 query
// rows per block (32 per wave as 2 M-tiles -> two independent WMMA chains),
// keys processed in chunks of 32.
// ---------------------------------------------------------------------------
#define VSTR 40   // LDS Vt row stride in halfs (padded; 80B rows, 16B aligned)

__global__ __launch_bounds__(128)
void flash_attn_kernel(const __bf16* __restrict__ qkv, __bf16* __restrict__ out) {
  __shared__ __bf16 Vt[64 * VSTR];       // V^T staging: [d][key] for one chunk
  __shared__ __bf16 Pst[4 * 32 * 32];    // per-wave P tile (32x32)

  const int lane = threadIdx.x & 31;
  const int wave = threadIdx.x >> 5;
  const int bh = blockIdx.y;             // 0..31
  const int b = bh >> 4;
  const int h = bh & 15;
  const int q0 = blockIdx.x * 128 + wave * 32;
  const int l15 = lane & 15;
  const int hiHalf = lane >> 4;
  const int koA = hiHalf ? 16 : 0;
  const int koB = hiHalf ? 16 : 0;

  const size_t rs = QKV_N;               // qkv row stride (halfs)
  const __bf16* Qbase = qkv + (size_t)b * SEQ * rs + (size_t)h * DK;
  const __bf16* Kbase = Qbase + D_MODEL;
  const __bf16* Vbase = Qbase + 2 * D_MODEL;

  // Q fragments: 2 M-tiles x 2 dk-slices, loaded once.
  v16bf aq[2][2];
#pragma unroll
  for (int mi = 0; mi < 2; ++mi) {
    const __bf16* qrow = Qbase + (size_t)(q0 + mi * 16 + l15) * rs;
#pragma unroll
    for (int j = 0; j < 2; ++j) {
      const __bf16* p = qrow + j * 32 + koA;
      ((v8bf*)&aq[mi][j])[0] = *(const v8bf*)(p);
      ((v8bf*)&aq[mi][j])[1] = *(const v8bf*)(p + 8);
    }
  }

  v8f o[2][4] = {};
  float mrun[2][8], lrun[2][8];
#pragma unroll
  for (int mi = 0; mi < 2; ++mi)
#pragma unroll
    for (int r = 0; r < 8; ++r) { mrun[mi][r] = -INFINITY; lrun[mi][r] = 0.0f; }

  __bf16* Pl = Pst + wave * 32 * 32;
  const float scale = 0.125f;            // 1/sqrt(64)

  for (int kc = 0; kc < SEQ; kc += 32) {
    __syncthreads();                     // Vt safe to overwrite
    {
      // Stage V chunk transposed: thread loads V[key][d0..d0+15], scatters.
      const int key = threadIdx.x >> 2;            // 0..31
      const int d0 = (threadIdx.x & 3) * 16;
      const __bf16* vrow = Vbase + (size_t)(kc + key) * rs + d0;
      v8bf v0 = *(const v8bf*)(vrow);
      v8bf v1 = *(const v8bf*)(vrow + 8);
#pragma unroll
      for (int i = 0; i < 8; ++i) {
        Vt[(d0 + i) * VSTR + key] = v0[i];
        Vt[(d0 + 8 + i) * VSTR + key] = v1[i];
      }
    }
    __syncthreads();                     // Vt visible to all waves

    // S = Q @ K^T for 32 keys; K B-frags shared by both M-tiles.
    v8f sc[2][2] = {};
#pragma unroll
    for (int nt = 0; nt < 2; ++nt) {
      const __bf16* krow = Kbase + (size_t)(kc + nt * 16 + l15) * rs;
      v16bf bk0, bk1;
      ((v8bf*)&bk0)[0] = *(const v8bf*)(krow + koB);
      ((v8bf*)&bk0)[1] = *(const v8bf*)(krow + koB + 8);
      ((v8bf*)&bk1)[0] = *(const v8bf*)(krow + 32 + koB);
      ((v8bf*)&bk1)[1] = *(const v8bf*)(krow + 32 + koB + 8);
#pragma unroll
      for (int mi = 0; mi < 2; ++mi)
        sc[mi][nt] = wmma_bf16(aq[mi][0], bk0, sc[mi][nt]);
#pragma unroll
      for (int mi = 0; mi < 2; ++mi)
        sc[mi][nt] = wmma_bf16(aq[mi][1], bk1, sc[mi][nt]);
    }

    // Online softmax; write P (bf16) into per-wave LDS region (32x32).
#pragma unroll
    for (int mi = 0; mi < 2; ++mi) {
#pragma unroll
      for (int r = 0; r < 8; ++r) {
        float x0 = sc[mi][0][r] * scale;
        float x1 = sc[mi][1][r] * scale;
        float mx = redmax16(fmaxf(x0, x1));
        float mnew = fmaxf(mrun[mi][r], mx);
        float alpha = __expf(mrun[mi][r] - mnew);
        float p0 = __expf(x0 - mnew);
        float p1 = __expf(x1 - mnew);
        float rsum = redsum16(p0 + p1);
        lrun[mi][r] = lrun[mi][r] * alpha + rsum;
        mrun[mi][r] = mnew;
#pragma unroll
        for (int t = 0; t < 4; ++t) o[mi][t][r] *= alpha;
        const int m = mi * 16 + r + (hiHalf ? 8 : 0);
        Pl[m * 32 + l15] = (__bf16)p0;
        Pl[m * 32 + 16 + l15] = (__bf16)p1;
      }
    }
    // Per-wave LDS restage (C-layout -> A-layout); enforce store->load order.
    asm volatile("s_wait_dscnt 0" ::: "memory");

    v16bf ap[2];
#pragma unroll
    for (int mi = 0; mi < 2; ++mi) {
      const __bf16* p = Pl + (mi * 16 + l15) * 32 + koA;
      ((v8bf*)&ap[mi])[0] = *(const v8bf*)(p);
      ((v8bf*)&ap[mi])[1] = *(const v8bf*)(p + 8);
    }
    // O += P @ V  (4 d-tiles of 16); V B-frags shared by both M-tiles.
#pragma unroll
    for (int t = 0; t < 4; ++t) {
      v16bf bv;
      const __bf16* p = Vt + (t * 16 + l15) * VSTR + koB;
      ((v8bf*)&bv)[0] = *(const v8bf*)(p);
      ((v8bf*)&bv)[1] = *(const v8bf*)(p + 8);
#pragma unroll
      for (int mi = 0; mi < 2; ++mi)
        o[mi][t] = wmma_bf16(ap[mi], bv, o[mi][t]);
    }
  }

  // Normalize and store bf16 attention output [NTOK, D_MODEL].
#pragma unroll
  for (int mi = 0; mi < 2; ++mi) {
#pragma unroll
    for (int r = 0; r < 8; ++r) {
      const float inv = 1.0f / lrun[mi][r];
      const int row = q0 + mi * 16 + r + (hiHalf ? 8 : 0);
      const size_t g = ((size_t)b * SEQ + row) * D_MODEL + (size_t)h * DK;
#pragma unroll
      for (int t = 0; t < 4; ++t)
        out[g + t * 16 + l15] = (__bf16)(o[mi][t][r] * inv);
    }
  }
}

// ---------------------------------------------------------------------------
extern "C" void kernel_launch(void* const* d_in, const int* in_sizes, int n_in,
                              void* d_out, int out_size, void* d_ws, size_t ws_size,
                              hipStream_t stream) {
  const float* x      = (const float*)d_in[0];  // [2,2048,1024]
  const float* W_qkv  = (const float*)d_in[1];  // [3072,1024]
  const float* W_proj = (const float*)d_in[2];  // [1024,1024]
  const float* b_proj = (const float*)d_in[3];  // [1024]

  // Workspace layout (bf16 halfs): ~48 MB total.
  __bf16* xb     = (__bf16*)d_ws;
  __bf16* wqkvb  = xb     + (size_t)NTOK * D_MODEL;       // 4096*1024
  __bf16* wprojb = wqkvb  + (size_t)QKV_N * D_MODEL;      // 3072*1024
  __bf16* qkvb   = wprojb + (size_t)D_MODEL * D_MODEL;    // 1024*1024
  __bf16* attnb  = qkvb   + (size_t)NTOK * QKV_N;         // 4096*3072

  // 1) fp32 -> bf16 conversions
  {
    int n;
    n = NTOK * D_MODEL;
    cvt_f32_bf16_kernel<<<(n + 255) / 256, 256, 0, stream>>>(x, xb, n);
    n = QKV_N * D_MODEL;
    cvt_f32_bf16_kernel<<<(n + 255) / 256, 256, 0, stream>>>(W_qkv, wqkvb, n);
    n = D_MODEL * D_MODEL;
    cvt_f32_bf16_kernel<<<(n + 255) / 256, 256, 0, stream>>>(W_proj, wprojb, n);
  }

  // 2) QKV projection: [4096,3072] = xb @ W_qkv^T (bf16 out)
  {
    dim3 grid(QKV_N / 128, NTOK / 128);
    wmma_gemm_kernel<true><<<grid, 256, 0, stream>>>(
        xb, wqkvb, (void*)qkvb, nullptr, NTOK, QKV_N, D_MODEL);
  }

  // 3) Flash attention per (batch, head): 128 queries per block
  {
    dim3 grid(SEQ / 128, BATCH * N_HEADS);
    flash_attn_kernel<<<grid, 128, 0, stream>>>(qkvb, attnb);
  }

  // 4) Output projection + bias: [4096,1024] f32 -> d_out
  {
    dim3 grid(D_MODEL / 128, NTOK / 128);
    wmma_gemm_kernel<false><<<grid, 256, 0, stream>>>(
        attnb, wprojb, d_out, b_proj, NTOK, D_MODEL, D_MODEL);
  }
}
